// GLAN_41729902247977
// MI455X (gfx1250) — matrix operand order repaired and checked by reading, compiled, not run on hardware
//
#include <hip/hip_runtime.h>
#include <hip/hip_bf16.h>

typedef __attribute__((ext_vector_type(16))) _Float16 v16h;
typedef __attribute__((ext_vector_type(8)))  float    v8f;
typedef __attribute__((ext_vector_type(4)))  float    f4;
typedef __attribute__((ext_vector_type(4)))  _Float16 h4;

static constexpr int kN  = 50000;   // nodes
static constexpr int kE  = 600000;  // edges
static constexpr int kF  = 128;     // NFEAT == NHID
static constexpr int kC  = 64;      // NCLASS
static constexpr int kE0 = 2;       // EDIM0
static constexpr int kNB = 256;     // colstats partial blocks

__device__ __forceinline__ float relu_(float v) { return v > 0.f ? v : 0.f; }

// ---------------------------------------------------------------------------
// WMMA term: acc += A(16xK f16, LDS, row stride K) @ W_term (pre-swizzled
// fragments in global).  Fragment (kcIdx, nt) = 512 halves; lane L owns the
// 16 contiguous halves at [frag*512 + L*16] which are exactly its B operand:
//   B 32x16 f16 : lane L -> col N = L&15; half j -> K = 16*(L>>4)+j
// C / D layout per CDNA5 ISA 7.12.2; A uses this code's own K order (wave32):
//   A 16x32 f16 : lane L -> row M = L&15; half i -> K = 16*(L>>4)+i
//                 (matches B's K order)
//   C/D 16x16 f32: lane L -> col N = L&15; vgpr r -> row M = r + 8*(L>>4)
// ---------------------------------------------------------------------------
template <int K, int OUT>
__device__ __forceinline__ v8f wmma_term_sw(v8f acc,
    const _Float16* __restrict__ A, const _Float16* __restrict__ WB,
    int ntIdx, int lane)
{
  const int m  = lane & 15;
  const int hs = lane >> 4;
  constexpr int NT = OUT / 16;
#pragma unroll
  for (int kc = 0; kc < K; kc += 32) {
    v16h a;
#pragma unroll
    for (int i = 0; i < 8; ++i) {
      a[i]     = A[m * K + kc + hs * 16 + i];
      a[i + 8] = A[m * K + kc + hs * 16 + 8 + i];
    }
    const _Float16* bp = WB + (((size_t)(kc / 32) * NT + ntIdx) << 9) + lane * 16;
    v16h b = *(const v16h*)bp;
    acc = __builtin_amdgcn_wmma_f32_16x16x32_f16(false, a, false, b,
                                                 (short)0, acc, false, false);
  }
  return acc;
}

// ------------------------- small utility kernels ---------------------------
__global__ void zero_kernel(float* __restrict__ p, int n) {
  int i = blockIdx.x * blockDim.x + threadIdx.x;
  if (i < n) p[i] = 0.f;
}

__global__ void count_kernel(const int* __restrict__ e1, float* __restrict__ cnt, int E) {
  int i = blockIdx.x * blockDim.x + threadIdx.x;
  if (i < E) atomicAdd(&cnt[e1[i]], 1.0f);
}

// Swizzle rows [rowOff, rowOff+Krows) of fp32 W[ldw-major] into f16 WMMA
// B-fragments (see wmma_term_sw header comment).
__global__ void wswz_kernel(const float* __restrict__ W, int ldw, int rowOff,
                            int ntilesN, _Float16* __restrict__ WB, int total) {
  int idx = blockIdx.x * blockDim.x + threadIdx.x;
  if (idx >= total) return;
  int fid = idx >> 9;
  int pos = idx & 511;
  int L = pos >> 4, j = pos & 15;
  int kcIdx = fid / ntilesN, nt = fid - kcIdx * ntilesN;
  int Kr = kcIdx * 32 + ((L >> 4) << 4) + j;
  int N  = (nt << 4) + (L & 15);
  WB[idx] = (_Float16)W[(size_t)(rowOff + Kr) * ldw + N];
}

// Coalesced two-stage column stats: stage 1 -> per-block partial max/min/sum.
template <int COLS, typename T>
__global__ void colstats_part(const T* __restrict__ X, int rows,
                              float* __restrict__ part) {
  constexpr int RPB = 256 / COLS;          // rows per block-iteration
  const int tid = threadIdx.x;
  const int c  = tid & (COLS - 1);
  const int r0 = tid / COLS;
  float vmax = -3.402823466e38f, vmin = 3.402823466e38f, vsum = 0.f;
  for (int r = blockIdx.x * RPB + r0; r < rows; r += (int)gridDim.x * RPB) {
    float v = (float)X[(size_t)r * COLS + c];
    vmax = fmaxf(vmax, v);
    vmin = fminf(vmin, v);
    vsum += v;
  }
  __shared__ float s0[256], s1[256], s2[256];
  s0[tid] = vmax; s1[tid] = vmin; s2[tid] = vsum;
  __syncthreads();
  for (int s = 128; s >= COLS; s >>= 1) {
    if (tid < s) {
      s0[tid] = fmaxf(s0[tid], s0[tid + s]);
      s1[tid] = fminf(s1[tid], s1[tid + s]);
      s2[tid] += s2[tid + s];
    }
    __syncthreads();
  }
  if (tid < COLS) {
    float* pb = part + (size_t)blockIdx.x * 3 * COLS;
    pb[c] = s0[tid]; pb[COLS + c] = s1[tid]; pb[2 * COLS + c] = s2[tid];
  }
}

// Stage 2: combine kNB partials -> out[3*cols] = [max | min | mean].
__global__ void colstats_final(const float* __restrict__ part, int nb, int cols,
                               int rows, float* __restrict__ out) {
  int idx = blockIdx.x * blockDim.x + threadIdx.x;
  if (idx >= 3 * cols) return;
  int cat = idx / cols, c = idx - cat * cols;
  float v = (cat == 0) ? -3.402823466e38f : (cat == 1 ? 3.402823466e38f : 0.f);
  for (int b = 0; b < nb; ++b) {
    float p = part[(size_t)b * 3 * cols + cat * cols + c];
    v = (cat == 0) ? fmaxf(v, p) : (cat == 1 ? fminf(v, p) : v + p);
  }
  out[idx] = (cat == 2) ? v / (float)rows : v;
}

// out[t] = relu(b[t] + sum_k s[k] * W[k, t]) ; 1-row MLP (kappa_v / kappa_e)
__global__ void ctx_kernel(const float* __restrict__ s, const float* __restrict__ W,
                           const float* __restrict__ b, float* __restrict__ out,
                           int K, int outdim) {
  int t = blockIdx.x * blockDim.x + threadIdx.x;
  if (t >= outdim) return;
  float acc = b[t];
  for (int k = 0; k < K; ++k) acc += s[k] * W[(size_t)k * outdim + t];
  out[t] = relu_(acc);
}

// w[e] = relu(tau_b + x[e0]·tauW[0:INV] + x[e1]·tauW[INV:2INV]) ; one wave/edge
template <int INV>
__global__ void tau_kernel(const float* __restrict__ X,
                           const int* __restrict__ e0, const int* __restrict__ e1,
                           const float* __restrict__ tauW, const float* __restrict__ taub,
                           float* __restrict__ w, int E) {
  static_assert(INV == 128, "one f4 per lane");
  const int gw   = (int)((blockIdx.x * blockDim.x + threadIdx.x) >> 5);
  const int lane = threadIdx.x & 31;
  if (gw >= E) return;
  const int a = e0[gw], b = e1[gw];
  constexpr int INV4 = INV / 4;
  f4 xa = ((const f4*)X)[(size_t)a * INV4 + lane];
  f4 xb = ((const f4*)X)[(size_t)b * INV4 + lane];
  f4 ta = ((const f4*)tauW)[lane];
  f4 tb = ((const f4*)tauW)[INV4 + lane];
  float acc = xa[0]*ta[0] + xa[1]*ta[1] + xa[2]*ta[2] + xa[3]*ta[3]
            + xb[0]*tb[0] + xb[1]*tb[1] + xb[2]*tb[2] + xb[3]*tb[3];
  for (int m = 16; m >= 1; m >>= 1) acc += __shfl_xor(acc, m, 32);
  if (lane == 0) w[gw] = relu_(acc + taub[0]);
}

// ---------------------------------------------------------------------------
// Fused per-edge kernel: one 16-edge tile, shared LDS staging for
//   sVi = vi*cv, sVj = vj*cv, sAw = w*vi*cv, sE = ea*ce
// then
//   msg  = relu([e_trans | w*x_trans[e0]] @ pv1_W + b1) -> atomic scatter
//   eout = relu([vi*cv | vj*cv | ea*ce]   @ pe_W  + b2) -> coalesced store
// ---------------------------------------------------------------------------
template <int INV, int INE, int OUT, typename EaT, typename EoutT>
__global__ void __launch_bounds__(256) edge_kernel(
    const float* __restrict__ X, const EaT* __restrict__ EA,
    const int* __restrict__ e0, const int* __restrict__ e1,
    const float* __restrict__ cv, const float* __restrict__ ce,
    const float* __restrict__ wtau,
    const _Float16* __restrict__ WBmsgE, const _Float16* __restrict__ WBmsgX,
    const float* __restrict__ WfMsg, const float* __restrict__ biasMsg,
    const _Float16* __restrict__ WBpeA, const _Float16* __restrict__ WBpeB,
    const _Float16* __restrict__ WBpeC,
    const float* __restrict__ WfPe, const float* __restrict__ biasPe,
    float* __restrict__ ssum, EoutT* __restrict__ eout, int E)
{
  constexpr int INV4 = INV / 4;
  constexpr int OUT4 = OUT / 4;
  __shared__ alignas(16) _Float16 sVi[16 * INV], sVj[16 * INV], sAw[16 * INV];
  __shared__ alignas(16) _Float16 sE[(INE >= 32) ? 16 * INE : 16];
  __shared__ float sEt[(INE < 32) ? 16 * INE : 1];
  __shared__ alignas(16) float sOut[16 * OUT];
  __shared__ int sSrc[16], sDst[16];
  __shared__ float sWt[16];
  const int row0 = blockIdx.x * 16;
  const int tid = threadIdx.x, nthr = blockDim.x;

  if (tid < 16) {
    int e = row0 + tid; if (e >= E) e = E - 1;
    sSrc[tid] = e0[e]; sDst[tid] = e1[e]; sWt[tid] = wtau[e];
  }
  __syncthreads();

  for (int idx = tid; idx < 16 * INV4; idx += nthr) {
    int r = idx / INV4, kv = idx - r * INV4;
    f4 cvv = ((const f4*)cv)[kv];
    f4 xi = ((const f4*)X)[(size_t)sSrc[r] * INV4 + kv];
    f4 xj = ((const f4*)X)[(size_t)sDst[r] * INV4 + kv];
    float wt = sWt[r];
    f4 pi, pj, pw;
#pragma unroll
    for (int q = 0; q < 4; ++q) {
      pi[q] = xi[q] * cvv[q];
      pj[q] = xj[q] * cvv[q];
      pw[q] = pi[q] * wt;
    }
    ((h4*)sVi)[idx] = __builtin_convertvector(pi, h4);
    ((h4*)sVj)[idx] = __builtin_convertvector(pj, h4);
    ((h4*)sAw)[idx] = __builtin_convertvector(pw, h4);
  }
  if constexpr (INE >= 32) {
    constexpr int INE4 = INE / 4;
    for (int idx = tid; idx < 16 * INE4; idx += nthr) {
      int r = idx / INE4, kv = idx - r * INE4;
      int e = row0 + r; if (e >= E) e = E - 1;
      f4 ev;
      if constexpr (sizeof(EaT) == 2) {
        h4 ev4 = ((const h4*)EA)[(size_t)e * INE4 + kv];
        ev = __builtin_convertvector(ev4, f4);
      } else {
        ev = ((const f4*)EA)[(size_t)e * INE4 + kv];
      }
      f4 cev = ((const f4*)ce)[kv];
      f4 p;
#pragma unroll
      for (int q = 0; q < 4; ++q) p[q] = ev[q] * cev[q];
      ((h4*)sE)[idx] = __builtin_convertvector(p, h4);
    }
  } else {
    for (int idx = tid; idx < 16 * INE; idx += nthr) {
      int r = idx / INE, k = idx - r * INE;
      int e = row0 + r; if (e >= E) e = E - 1;
      sEt[idx] = (float)EA[(size_t)e * INE + k] * ce[k];
    }
  }
  __syncthreads();

  const int lane = tid & 31;
  const int nt = tid >> 5;
  const int hs = lane >> 4;
  const int n  = nt * 16 + (lane & 15);

  // ---- message GEMM + atomic scatter (segment sum) ----
  {
    v8f acc = {0.f,0.f,0.f,0.f,0.f,0.f,0.f,0.f};
    if constexpr (INE >= 32) acc = wmma_term_sw<INE, OUT>(acc, sE, WBmsgE, nt, lane);
    acc = wmma_term_sw<INV, OUT>(acc, sAw, WBmsgX, nt, lane);
    const float bn = biasMsg[n];
    float wfm[(INE < 32) ? INE : 1];
    if constexpr (INE < 32) {
#pragma unroll
      for (int j = 0; j < INE; ++j) wfm[j] = WfMsg[(size_t)j * OUT + n];
    }
#pragma unroll
    for (int r = 0; r < 8; ++r) {
      int M = r + 8 * hs;
      int e = row0 + M;
      if (e < E) {
        float v = acc[r] + bn;
        if constexpr (INE < 32) {
#pragma unroll
          for (int j = 0; j < INE; ++j) v += sEt[M * INE + j] * wfm[j];
        }
        atomicAdd(&ssum[(size_t)sDst[M] * OUT + n], relu_(v));
      }
    }
  }

  // ---- edge-feature GEMM + coalesced store ----
  {
    v8f acc = {0.f,0.f,0.f,0.f,0.f,0.f,0.f,0.f};
    acc = wmma_term_sw<INV, OUT>(acc, sVi, WBpeA, nt, lane);
    acc = wmma_term_sw<INV, OUT>(acc, sVj, WBpeB, nt, lane);
    if constexpr (INE >= 32) acc = wmma_term_sw<INE, OUT>(acc, sE, WBpeC, nt, lane);
    const float bn = biasPe[n];
    float wfp[(INE < 32) ? INE : 1];
    if constexpr (INE < 32) {
#pragma unroll
      for (int j = 0; j < INE; ++j) wfp[j] = WfPe[(size_t)(2 * INV + j) * OUT + n];
    }
#pragma unroll
    for (int r = 0; r < 8; ++r) {
      int M = r + 8 * hs;
      float v = acc[r] + bn;
      if constexpr (INE < 32) {
#pragma unroll
        for (int j = 0; j < INE; ++j) v += sEt[M * INE + j] * wfp[j];
      }
      sOut[M * OUT + n] = relu_(v);
    }
  }
  __syncthreads();

  for (int idx = tid; idx < 16 * OUT4; idx += nthr) {
    int r = idx / OUT4;
    int e = row0 + r;
    if (e < E) {
      f4 v = ((const f4*)sOut)[idx];
      if constexpr (sizeof(EoutT) == 2)
        ((h4*)eout)[(size_t)e * OUT4 + (idx - r * OUT4)] = __builtin_convertvector(v, h4);
      else
        ((f4*)eout)[(size_t)e * OUT4 + (idx - r * OUT4)] = v;
    }
  }
}

// ---------------------------------------------------------------------------
// x_out = relu([x | ssum/max(cnt,1)] @ pv2_W + b)  (coalesced f4 stores)
// ---------------------------------------------------------------------------
template <int INV, int OUT>
__global__ void __launch_bounds__(256) pv2_kernel(
    const float* __restrict__ X, const float* __restrict__ ssum,
    const float* __restrict__ cnt,
    const _Float16* __restrict__ WBx, const _Float16* __restrict__ WBg,
    const float* __restrict__ bias,
    float* __restrict__ xout, int Nn)
{
  constexpr int INV4 = INV / 4, OUT4 = OUT / 4;
  __shared__ alignas(16) _Float16 sA[16 * INV], sG[16 * OUT];
  __shared__ alignas(16) float sOut[16 * OUT];
  const int row0 = blockIdx.x * 16;
  const int tid = threadIdx.x, nthr = blockDim.x;

  for (int idx = tid; idx < 16 * INV4; idx += nthr) {
    int r = idx / INV4, kv = idx - r * INV4;
    int node = row0 + r; if (node >= Nn) node = Nn - 1;
    f4 xv = ((const f4*)X)[(size_t)node * INV4 + kv];
    ((h4*)sA)[idx] = __builtin_convertvector(xv, h4);
  }
  for (int idx = tid; idx < 16 * OUT4; idx += nthr) {
    int r = idx / OUT4, kv = idx - r * OUT4;
    int node = row0 + r; if (node >= Nn) node = Nn - 1;
    float inv = 1.f / fmaxf(cnt[node], 1.f);
    f4 sv = ((const f4*)ssum)[(size_t)node * OUT4 + kv];
    f4 p;
#pragma unroll
    for (int q = 0; q < 4; ++q) p[q] = sv[q] * inv;
    ((h4*)sG)[idx] = __builtin_convertvector(p, h4);
  }
  __syncthreads();

  const int lane = tid & 31;
  const int nt = tid >> 5;
  v8f acc = {0.f,0.f,0.f,0.f,0.f,0.f,0.f,0.f};
  acc = wmma_term_sw<INV, OUT>(acc, sA, WBx, nt, lane);
  acc = wmma_term_sw<OUT, OUT>(acc, sG, WBg, nt, lane);

  const int hs = lane >> 4;
  const int n  = nt * 16 + (lane & 15);
  const float bn = bias[n];
#pragma unroll
  for (int r = 0; r < 8; ++r) {
    int M = r + 8 * hs;
    sOut[M * OUT + n] = relu_(acc[r] + bn);
  }
  __syncthreads();

  for (int idx = tid; idx < 16 * OUT4; idx += nthr) {
    int r = idx / OUT4;
    int node = row0 + r;
    if (node < Nn)
      ((f4*)xout)[(size_t)node * OUT4 + (idx - r * OUT4)] = ((const f4*)sOut)[idx];
  }
}

// ------------------------------- host side ---------------------------------
static inline int cdiv(int a, int b) { return (a + b - 1) / b; }

struct SwzBufs { _Float16 *msgE, *msgX, *pv2X, *pv2G, *peA, *peB, *peC; };

template <int INV, int INE, int OUT, typename EaT, typename EoutT>
static void run_layer(hipStream_t stream,
    const float* X, const EaT* EA, const int* e0, const int* e1,
    const float* const* P,
    float* statsX, float* statsE, float* partbuf, float* cv, float* ce,
    float* wtau, float* ssum, const float* cnt,
    const SwzBufs& sb, float* xOut, EoutT* eOut)
{
  enum { KV_W, KV_B, KE_W, KE_B, PE_W, PE_B, TAU_W, TAU_B, PV1_W, PV1_B, PV2_W, PV2_B };

  colstats_part<INV, float><<<kNB, 256, 0, stream>>>(X, kN, partbuf);
  colstats_final<<<cdiv(3 * INV, 256), 256, 0, stream>>>(partbuf, kNB, INV, kN, statsX);
  colstats_part<INE, EaT><<<kNB, 256, 0, stream>>>(EA, kE, partbuf);
  colstats_final<<<cdiv(3 * INE, 256), 256, 0, stream>>>(partbuf, kNB, INE, kE, statsE);
  ctx_kernel<<<1, 128, 0, stream>>>(statsX, P[KV_W], P[KV_B], cv, 3 * INV, INV);
  ctx_kernel<<<1, 128, 0, stream>>>(statsE, P[KE_W], P[KE_B], ce, 3 * INE, INE);

  auto swz = [&](const float* W, int rowOff, int Krows, _Float16* dst) {
    int total = (Krows / 32) * (OUT / 16) * 512;
    wswz_kernel<<<cdiv(total, 256), 256, 0, stream>>>(W, OUT, rowOff, OUT / 16, dst, total);
  };
  if (INE >= 32) swz(P[PV1_W], 0, INE, sb.msgE);
  swz(P[PV1_W], INE, INV, sb.msgX);
  swz(P[PV2_W], 0, INV, sb.pv2X);
  swz(P[PV2_W], INV, OUT, sb.pv2G);
  swz(P[PE_W], 0, INV, sb.peA);
  swz(P[PE_W], INV, INV, sb.peB);
  if (INE >= 32) swz(P[PE_W], 2 * INV, INE, sb.peC);

  tau_kernel<INV><<<cdiv(kE, 8), 256, 0, stream>>>(X, e0, e1, P[TAU_W], P[TAU_B], wtau, kE);
  zero_kernel<<<cdiv(kN * OUT, 256), 256, 0, stream>>>(ssum, kN * OUT);

  constexpr int nthr = 32 * (OUT / 16);
  edge_kernel<INV, INE, OUT, EaT, EoutT><<<cdiv(kE, 16), nthr, 0, stream>>>(
      X, EA, e0, e1, cv, ce, wtau,
      sb.msgE, sb.msgX, P[PV1_W], P[PV1_B],
      sb.peA, sb.peB, sb.peC, P[PE_W], P[PE_B],
      ssum, eOut, kE);
  pv2_kernel<INV, OUT><<<cdiv(kN, 16), nthr, 0, stream>>>(X, ssum, cnt,
      sb.pv2X, sb.pv2G, P[PV2_B], xOut, kN);
}

extern "C" void kernel_launch(void* const* d_in, const int* in_sizes, int n_in,
                              void* d_out, int out_size, void* d_ws, size_t ws_size,
                              hipStream_t stream)
{
  (void)in_sizes; (void)n_in; (void)out_size; (void)ws_size;
  const float* x  = (const float*)d_in[0];
  const int*   ei = (const int*)d_in[1];   // int32 (jax default x64 off)
  const float* ea = (const float*)d_in[2];
  const int* e0 = ei;
  const int* e1 = ei + kE;

  const float* P0[12];
  const float* P1[12];
  for (int i = 0; i < 12; ++i) P0[i] = (const float*)d_in[3 + i];
  for (int i = 0; i < 12; ++i) P1[i] = (const float*)d_in[15 + i];

  char* ws = (char*)d_ws;
  size_t off = 0;
  auto take = [&](size_t bytes) -> char* {
    char* p = ws + off;
    off += (bytes + 255) & ~(size_t)255;
    return p;
  };
  float*    x1     = (float*)take((size_t)kN * kF * 4);
  _Float16* ea1    = (_Float16*)take((size_t)kE * kF * 2);   // f16 intermediate
  float*    ssum   = (float*)take((size_t)kN * kF * 4);
  float*    cnt    = (float*)take((size_t)kN * 4);
  float*    wtau   = (float*)take((size_t)kE * 4);
  float*    part   = (float*)take((size_t)kNB * 3 * kF * 4);
  float*    statsX = (float*)take(3 * kF * 4);
  float*    statsE = (float*)take(3 * kF * 4);
  float*    cv     = (float*)take(kF * 4);
  float*    ce     = (float*)take(kF * 4);

  // swizzled weight fragments (max (128/32)*(128/16)*512 halves = 32 KB each)
  auto takeWB = [&]() { return (_Float16*)take(4 * 8 * 512 * 2); };
  SwzBufs sb0{nullptr,  takeWB(), takeWB(), takeWB(), takeWB(), takeWB(), nullptr};
  SwzBufs sb1{takeWB(), takeWB(), takeWB(), takeWB(), takeWB(), takeWB(), takeWB()};

  // In-degree counts (identical for both layers)
  zero_kernel<<<cdiv(kN, 256), 256, 0, stream>>>(cnt, kN);
  count_kernel<<<cdiv(kE, 256), 256, 0, stream>>>(e1, cnt, kE);

  float* x2  = (float*)d_out;                 // [N, NCLASS]
  float* ea2 = x2 + (size_t)kN * kC;          // [E, NCLASS]

  // Layer 0: in_v=128, in_e=2, out=128 ; edge features stored f16
  run_layer<kF, kE0, kF, float, _Float16>(stream, x, ea, e0, e1, P0,
      statsX, statsE, part, cv, ce, wtau, ssum, cnt, sb0, x1, ea1);
  // Layer 1: in_v=128, in_e=128, out=64 ; final outputs fp32
  run_layer<kF, kF, kC, _Float16, float>(stream, x1, ea1, e0, e1, P1,
      statsX, statsE, part, cv, ce, wtau, ssum, cnt, sb1, x2, ea2);
}
